// JANET_24910810317338
// MI455X (gfx1250) — matrix operand order repaired and checked by reading, not compiled, not run on hardware
//
#include <hip/hip_runtime.h>

#define T_STEPS 512
#define BATCH   64
#define DIN     512
#define HID     1024
#define BH      (BATCH * HID)   // 65536 elements per timestep slice

typedef __attribute__((ext_vector_type(16))) _Float16 v16h;
typedef __attribute__((ext_vector_type(8)))  _Float16 v8h;
typedef __attribute__((ext_vector_type(8)))  float    v8f;

static __device__ __forceinline__ v16h cat16(v8h lo, v8h hi) {
    union { v16h v; v8h h[2]; } u;
    u.h[0] = lo; u.h[1] = hi;
    return u.v;
}

static __device__ __forceinline__ float fsigmoid(float x) {
    return 1.0f / (1.0f + __expf(-x));
}
static __device__ __forceinline__ float ftanh(float x) {
    float e = __expf(-2.0f * x);
    return (1.0f - e) / (1.0f + e);
}

// ---------------------------------------------------------------------------
// f32 -> f16 conversion (weights / activations), grid-stride
// ---------------------------------------------------------------------------
__global__ void cvt_f32_f16(const float* __restrict__ s, _Float16* __restrict__ d, int n) {
    int i = blockIdx.x * blockDim.x + threadIdx.x;
    int stride = gridDim.x * blockDim.x;
    for (; i < n; i += stride) d[i] = (_Float16)s[i];
}

__global__ void zero_f16(_Float16* __restrict__ d, int n) {
    int i = blockIdx.x * blockDim.x + threadIdx.x;
    if (i < n) d[i] = (_Float16)0.0f;
}

// ---------------------------------------------------------------------------
// Input-projection GEMM: P{f,g}[M,HID] = A[M,K] * W{f,g}[HID,K]^T + bias
//   block = 256 threads = 8 waves; each wave: one 16x16 tile for BOTH gates
//   grid = (HID/16, M/128)
// A fragment, contiguous k order (k = 16*hx + s, same as B): lane (m + 16*hx):
//   elems 0..15 = A[m][k0 + 16*hx .. +15]
// B fragment (32x16, B[k][n] = W[nbase+n][k]): lane (n + 16*hx):
//   elems 0..15 = W[nbase+n][k0 + 16*hx .. +15]  (one contiguous 32B run)
// C/D: lane (n + 16*hx), vgpr r -> (M = r + 8*hx, N = n)
// ---------------------------------------------------------------------------
__global__ __launch_bounds__(256)
void janet_proj(const _Float16* __restrict__ A, int K,
                const _Float16* __restrict__ Wf, const _Float16* __restrict__ Wg,
                const float* __restrict__ bf, const float* __restrict__ bg,
                float* __restrict__ Pf, float* __restrict__ Pg, float fsub)
{
    const int lane  = threadIdx.x & 31;
    const int wave  = threadIdx.x >> 5;
    const int mbase = blockIdx.y * 128 + wave * 16;
    const int nbase = blockIdx.x * 16;
    const int m     = lane & 15;
    const int hx    = lane >> 4;

    const _Float16* arow  = A  + (size_t)(mbase + m) * K;
    const _Float16* wfrow = Wf + (size_t)(nbase + m) * K;
    const _Float16* wgrow = Wg + (size_t)(nbase + m) * K;

    v8f accf = {}; v8f accg = {};
    for (int k0 = 0; k0 < K; k0 += 32) {
        v8h alo = *(const v8h*)(arow + k0 + hx * 16);
        v8h ahi = *(const v8h*)(arow + k0 + hx * 16 + 8);
        v16h a  = cat16(alo, ahi);
        v16h bfm = *(const v16h*)(wfrow + k0 + hx * 16);
        v16h bgm = *(const v16h*)(wgrow + k0 + hx * 16);
        accf = __builtin_amdgcn_wmma_f32_16x16x32_f16(false, a, false, bfm,
                                                      (short)0, accf, false, false);
        accg = __builtin_amdgcn_wmma_f32_16x16x32_f16(false, a, false, bgm,
                                                      (short)0, accg, false, false);
    }

    const int col = nbase + m;
    const float bfv = bf[col] + fsub;   // fsub = -BETA folded into f-gate bias
    const float bgv = bg[col];
#pragma unroll
    for (int r = 0; r < 8; ++r) {
        size_t idx = (size_t)(mbase + r + hx * 8) * HID + col;
        Pf[idx] = accf[r] + bfv;
        Pg[idx] = accg[r] + bgv;
    }
}

// ---------------------------------------------------------------------------
// Fused recurrent step:
//   F = sigmoid(Pf + h*Wf^T + bf); G = tanh(Pg + h*Wg^T + bg)
//   h' = F*h + (1-F)*G
//
// block = 256 (8 waves) covers 64 rows x 32 cols of h'; grid.x = HID/32 = 32.
// Rows 0..31 of h (32x1024 f16 = 64 KB) are staged into LDS once per step
// (CDNA5 WGP has 320 KB LDS; 132 KB/block -> 2 blocks per WGP). One barrier
// total; the K loop then runs barrier-free out of LDS.
// LDS row stride = 1032 halfs (2064 B): 16B-aligned rows, and the 16-lane
// strided A-fragment reads land on distinct banks (516 dwords = 4 mod 64).
// ---------------------------------------------------------------------------
#define LDS_ROW 1032
#define STEP_LDS_BYTES (BATCH * LDS_ROW * (int)sizeof(_Float16))   // 132096

template <bool WF32>
__global__ __launch_bounds__(256)
void janet_step(const float* __restrict__ Pf, const float* __restrict__ Pg,
                const _Float16* __restrict__ hprev,
                const _Float16* __restrict__ Wf, const _Float16* __restrict__ Wg,
                const float* __restrict__ bf, const float* __restrict__ bg,
                _Float16* __restrict__ hnew, float* __restrict__ out)
{
    extern __shared__ __align__(16) char smem[];
    _Float16* sH = (_Float16*)smem;

    const int tid  = threadIdx.x;
    const int lane = tid & 31;
    const int wave = tid >> 5;
    const int mbase = (wave & 3) * 16;                          // 4 row tiles
    const int nbase = ((blockIdx.x << 1) + (wave >> 2)) * 16;   // 2 col tiles/blk
    const int m  = lane & 15;
    const int hx = lane >> 4;

    // ---- stage h rows 0..31 [32 x 1024] into LDS: 4096 8-half segments ----
#pragma unroll
    for (int i = 0; i < 16; ++i) {
        int s    = tid + i * 256;        // 0..4095
        int row  = s >> 7;               // 0..31
        int segc = (s & 127) * 8;        // 0..1016, 16B granules
        *(v8h*)(&sH[row * LDS_ROW + segc]) =
            *(const v8h*)(hprev + row * HID + segc);
    }
    __syncthreads();

    const _Float16* wfrow = Wf + (size_t)(nbase + m) * HID;
    const _Float16* wgrow = Wg + (size_t)(nbase + m) * HID;
    const _Float16* sa0   = &sH[(mbase + m) * LDS_ROW + hx * 16];

    v8f accf = {}; v8f accg = {};
#pragma unroll 4
    for (int k0 = 0; k0 < HID; k0 += 32) {
        v16h a   = cat16(*(const v8h*)(sa0 + k0), *(const v8h*)(sa0 + k0 + 8));
        v16h bfm = *(const v16h*)(wfrow + k0 + hx * 16);
        v16h bgm = *(const v16h*)(wgrow + k0 + hx * 16);
        accf = __builtin_amdgcn_wmma_f32_16x16x32_f16(false, a, false, bfm,
                                                      (short)0, accf, false, false);
        accg = __builtin_amdgcn_wmma_f32_16x16x32_f16(false, a, false, bgm,
                                                      (short)0, accg, false, false);
    }

    const int col = nbase + m;
    const float bfv = bf[col];
    const float bgv = bg[col];
#pragma unroll
    for (int r = 0; r < 8; ++r) {
        int row = mbase + r + hx * 8;
        int idx = row * HID + col;
        float F  = fsigmoid(Pf[idx] + accf[r] + bfv);
        float G  = ftanh(Pg[idx] + accg[r] + bgv);
        float ho = (float)sH[row * LDS_ROW + col];   // h_old from LDS
        float hn = F * ho + (1.0f - F) * G;
        hnew[idx] = (_Float16)hn;
        if (WF32) out[idx] = hn;
    }
}

// ---------------------------------------------------------------------------
// Host-side orchestration (graph-capture safe: only kernel launches on stream)
// ---------------------------------------------------------------------------
extern "C" void kernel_launch(void* const* d_in, const int* in_sizes, int n_in,
                              void* d_out, int out_size, void* d_ws, size_t ws_size,
                              hipStream_t stream)
{
    (void)in_sizes; (void)n_in; (void)out_size; (void)ws_size;

    const float* X    = (const float*)d_in[0];
    const float* ifW0 = (const float*)d_in[1];  const float* ifB0 = (const float*)d_in[2];
    const float* hfW0 = (const float*)d_in[3];  const float* hfB0 = (const float*)d_in[4];
    const float* igW0 = (const float*)d_in[5];  const float* igB0 = (const float*)d_in[6];
    const float* hgW0 = (const float*)d_in[7];  const float* hgB0 = (const float*)d_in[8];
    const float* ifW1 = (const float*)d_in[9];  const float* ifB1 = (const float*)d_in[10];
    const float* hfW1 = (const float*)d_in[11]; const float* hfB1 = (const float*)d_in[12];
    const float* igW1 = (const float*)d_in[13]; const float* igB1 = (const float*)d_in[14];
    const float* hgW1 = (const float*)d_in[15]; const float* hgB1 = (const float*)d_in[16];
    float* Y1 = (float*)d_out;

    const size_t M0 = (size_t)T_STEPS * BATCH;   // 32768 rows

    // ---- workspace carving (256B aligned) ----
    char* ws = (char*)d_ws;
    size_t off = 0;
    auto take = [&](size_t bytes) -> char* {
        char* p = ws + off;
        off += (bytes + 255) & ~(size_t)255;
        return p;
    };
    _Float16* X16    = (_Float16*)take(M0 * DIN * sizeof(_Float16));
    _Float16* Y0_16  = (_Float16*)take((size_t)T_STEPS * BH * sizeof(_Float16));
    float*    Pf     = (float*)   take(M0 * HID * sizeof(float));
    float*    Pg     = (float*)   take(M0 * HID * sizeof(float));
    _Float16* ifW0h  = (_Float16*)take((size_t)HID * DIN * sizeof(_Float16));
    _Float16* igW0h  = (_Float16*)take((size_t)HID * DIN * sizeof(_Float16));
    _Float16* hfW0h  = (_Float16*)take((size_t)HID * HID * sizeof(_Float16));
    _Float16* hgW0h  = (_Float16*)take((size_t)HID * HID * sizeof(_Float16));
    _Float16* ifW1h  = (_Float16*)take((size_t)HID * HID * sizeof(_Float16));
    _Float16* igW1h  = (_Float16*)take((size_t)HID * HID * sizeof(_Float16));
    _Float16* hfW1h  = (_Float16*)take((size_t)HID * HID * sizeof(_Float16));
    _Float16* hgW1h  = (_Float16*)take((size_t)HID * HID * sizeof(_Float16));
    _Float16* hzero  = (_Float16*)take((size_t)BH * sizeof(_Float16));
    _Float16* h1a    = (_Float16*)take((size_t)BH * sizeof(_Float16));
    _Float16* h1b    = (_Float16*)take((size_t)BH * sizeof(_Float16));

    // ---- precision conversion ----
    auto cvt = [&](const float* s, _Float16* d, size_t n) {
        int blocks = (int)((n + 255) / 256);
        if (blocks > 65535) blocks = 65535;
        cvt_f32_f16<<<blocks, 256, 0, stream>>>(s, d, (int)n);
    };
    cvt(X,    X16,   M0 * DIN);
    cvt(ifW0, ifW0h, (size_t)HID * DIN);
    cvt(igW0, igW0h, (size_t)HID * DIN);
    cvt(hfW0, hfW0h, (size_t)HID * HID);
    cvt(hgW0, hgW0h, (size_t)HID * HID);
    cvt(ifW1, ifW1h, (size_t)HID * HID);
    cvt(igW1, igW1h, (size_t)HID * HID);
    cvt(hfW1, hfW1h, (size_t)HID * HID);
    cvt(hgW1, hgW1h, (size_t)HID * HID);
    zero_f16<<<BH / 256, 256, 0, stream>>>(hzero, BH);

    dim3 projGrid(HID / 16, (unsigned)(M0 / 128));

    // ---- layer 0 ----
    janet_proj<<<projGrid, 256, 0, stream>>>(X16, DIN, ifW0h, igW0h,
                                             ifB0, igB0, Pf, Pg, -1.0f /* -BETA */);
    for (int t = 0; t < T_STEPS; ++t) {
        const _Float16* hp = (t == 0) ? hzero : (Y0_16 + (size_t)(t - 1) * BH);
        janet_step<false><<<HID / 32, 256, STEP_LDS_BYTES, stream>>>(
            Pf + (size_t)t * BH, Pg + (size_t)t * BH, hp,
            hfW0h, hgW0h, hfB0, hgB0,
            Y0_16 + (size_t)t * BH, nullptr);
    }

    // ---- layer 1 ----
    janet_proj<<<projGrid, 256, 0, stream>>>(Y0_16, HID, ifW1h, igW1h,
                                             ifB1, igB1, Pf, Pg, -1.0f);
    for (int t = 0; t < T_STEPS; ++t) {
        const _Float16* hp = (t == 0) ? hzero : ((t - 1) & 1 ? h1b : h1a);
        _Float16*       hn = (t & 1) ? h1b : h1a;
        janet_step<true><<<HID / 32, 256, STEP_LDS_BYTES, stream>>>(
            Pf + (size_t)t * BH, Pg + (size_t)t * BH, hp,
            hfW1h, hgW1h, hfB1, hgB1,
            hn, Y1 + (size_t)t * BH);
    }
}
